// TSIM_5446018531502
// MI455X (gfx1250) — matrix operand order repaired and checked by reading, not compiled, not run on hardware
//
#include <hip/hip_runtime.h>

#define HW   14400
#define CC   256
#define DD   64
#define NB   8
#define BNEPS 1e-5f

typedef unsigned short u16;
typedef unsigned int   u32;
typedef __attribute__((ext_vector_type(16))) __bf16 bf16x16;
typedef __attribute__((ext_vector_type(8)))  float  f32x8;

__device__ __forceinline__ u16 f2bf(float f) {
    u32 u = __float_as_uint(f);
    u32 r = u + 0x7FFFu + ((u >> 16) & 1u);   // round-to-nearest-even
    return (u16)(r >> 16);
}
__device__ __forceinline__ float bf2f(u16 h) { return __uint_as_float(((u32)h) << 16); }

// ---------------------------------------------------------------------------
// Fold BN (eval) into 1x1 conv weights, convert to bf16.
//   w'[o,c] = w[o,c] * g[o]*rsqrt(v[o]+eps)
//   b'[o]   = (b[o]-m[o]) * g[o]*rsqrt(v[o]+eps) + bt[o]
// wqkv rows: [0,64)=q  [64,128)=k  [128,192)=v
// ---------------------------------------------------------------------------
__global__ void prep_kernel(
    const float* __restrict__ w_pp, const float* __restrict__ b_pp,
    const float* __restrict__ g_pp, const float* __restrict__ bt_pp,
    const float* __restrict__ m_pp, const float* __restrict__ v_pp,
    const float* __restrict__ w_q,  const float* __restrict__ b_q,
    const float* __restrict__ g_q,  const float* __restrict__ bt_q,
    const float* __restrict__ m_q,  const float* __restrict__ v_q,
    const float* __restrict__ w_k,  const float* __restrict__ b_k,
    const float* __restrict__ g_k,  const float* __restrict__ bt_k,
    const float* __restrict__ m_k,  const float* __restrict__ v_k,
    const float* __restrict__ w_v,  const float* __restrict__ b_v,
    const float* __restrict__ g_v,  const float* __restrict__ bt_v,
    const float* __restrict__ m_v,  const float* __restrict__ v_v,
    const float* __restrict__ w_out, const float* __restrict__ b_out,
    u16* __restrict__ wpp, float* __restrict__ bpp,
    u16* __restrict__ wqkv, float* __restrict__ bqkv,
    u16* __restrict__ wout, float* __restrict__ bout)
{
    int idx = blockIdx.x * 256 + threadIdx.x;
    if (idx < 65536) {                               // pp: 256x256
        int o = idx >> 8, c = idx & 255;
        float s = g_pp[o] * rsqrtf(v_pp[o] + BNEPS);
        wpp[idx] = f2bf(w_pp[idx] * s);
        if (c == 0) bpp[o] = (b_pp[o] - m_pp[o]) * s + bt_pp[o];
    } else if (idx < 65536 + 16384) {                // q: 64x256
        int j = idx - 65536; int o = j >> 8, c = j & 255;
        float s = g_q[o] * rsqrtf(v_q[o] + BNEPS);
        wqkv[j] = f2bf(w_q[j] * s);
        if (c == 0) bqkv[o] = (b_q[o] - m_q[o]) * s + bt_q[o];
    } else if (idx < 65536 + 32768) {                // k
        int j = idx - (65536 + 16384); int o = j >> 8, c = j & 255;
        float s = g_k[o] * rsqrtf(v_k[o] + BNEPS);
        wqkv[16384 + j] = f2bf(w_k[j] * s);
        if (c == 0) bqkv[64 + o] = (b_k[o] - m_k[o]) * s + bt_k[o];
    } else if (idx < 65536 + 49152) {                // v
        int j = idx - (65536 + 32768); int o = j >> 8, c = j & 255;
        float s = g_v[o] * rsqrtf(v_v[o] + BNEPS);
        wqkv[32768 + j] = f2bf(w_v[j] * s);
        if (c == 0) bqkv[128 + o] = (b_v[o] - m_v[o]) * s + bt_v[o];
    } else if (idx < 65536 + 49152 + 16384) {        // out: 256x64, no BN
        int j = idx - (65536 + 49152); int o = j >> 6, d = j & 63;
        wout[j] = f2bf(w_out[j]);
        if (d == 0) bout[o] = b_out[o];
    }
}

// f32 -> bf16 bulk convert (float4 -> ushort4)
__global__ void cvt_bf16(const float* __restrict__ x, u16* __restrict__ y, int n4)
{
    int i = blockIdx.x * 256 + threadIdx.x;
    if (i >= n4) return;
    float4 f = ((const float4*)x)[i];
    ushort4 o;
    o.x = f2bf(f.x); o.y = f2bf(f.y); o.z = f2bf(f.z); o.w = f2bf(f.w);
    ((ushort4*)y)[i] = o;
}

// ---------------------------------------------------------------------------
// WMMA GEMM:  Y[b] = epi( A[b] (MxK, bf16, row-major, contiguous) *
//                         B[b] (KxN, bf16, row stride ldb) + bias )
// Workgroup: 256 threads = 8 waves (2 M-waves x 4 N-waves), tile 64(M)x64(N).
// Each wave owns 32x16: TWO A fragments sharing ONE B fragment
// (12 LDS dwords per WMMA).  The two A fragments are 16 LDS rows apart
// (16*132*4 = 8448 B), far beyond the ds_load_2addr offset window (1020 B),
// so the load/store optimizer can only fuse intra-fragment adjacent-dword
// pairs -- whose adjacent destination registers already match the WMMA
// operand layout (no v_mov re-gather).
// K staged fully in LDS.
// A LDS: [64][K/2] uint (pair-packed bf16), row stride K/2+4 (16B aligned,
//        bank-conflict-free fragment reads).
// B LDS: K-pair interleaved: BU[(k>>1)*64 + n] = {B[k][n], B[k+1][n]}
//        -> every fragment register is one ds_load_b32 (stride-64 pairs fuse).
// Fragment layouts per CDNA5 ISA 7.12.2 (16-bit A 16x32, B 32x16, f32 C/D).
// ---------------------------------------------------------------------------
enum { EPI_RELU_F32 = 0, EPI_RELU_BF16 = 1, EPI_PLAIN_F32 = 2, EPI_GATE = 3 };

template<int EPI>
__global__ __launch_bounds__(256) void gemm_wmma(
    const u16* __restrict__ A, long strideA,
    const u16* __restrict__ B, long strideB, int ldb,
    const float* __restrict__ bias,
    void* __restrict__ Y, long strideY, int ldy, int K,
    const float* __restrict__ pre0, const float* __restrict__ post0,
    float* __restrict__ Y2)
{
    __shared__ u32 smem[64 * 132 + 128 * 64];    // sized for K=256 (65 KB)
    const int t  = threadIdx.x;
    const int b  = blockIdx.z;
    const int m0 = blockIdx.y * 64;
    const int n0 = blockIdx.x * 64;
    const int K2  = K >> 1;
    const int K2p = K2 + 4;
    u32* AU = smem;
    u32* BU = smem + 64 * K2p;

    // ---- stage A panel (64 x K bf16, contiguous) ----
    {
        const uint4* src  = (const uint4*)(A + b * strideA + (long)m0 * K);
        const int    rowv = K2 >> 2;            // uint4 per row
        const int    tot  = 64 * rowv;
        for (int j = t; j < tot; j += 256) {
            int r = j / rowv, cv = j - r * rowv;
            *(uint4*)(AU + r * K2p + cv * 4) = src[j];
        }
    }
    // ---- stage B panel (K x 64), pair-interleaved ----
    {
        const u16* Bp   = B + b * strideB + n0;
        const int  jobs = K2 * 8;               // 8 n-chunks of 8 per k-pair
        for (int j = t; j < jobs; j += 256) {
            int kp = j >> 3, nc = j & 7;
            uint4 g0 = *(const uint4*)(Bp + (long)(2 * kp)     * ldb + nc * 8);
            uint4 g1 = *(const uint4*)(Bp + (long)(2 * kp + 1) * ldb + nc * 8);
            const u16* a8 = (const u16*)&g0;
            const u16* c8 = (const u16*)&g1;
            u32 o[8];
            #pragma unroll
            for (int i = 0; i < 8; i++) o[i] = (u32)a8[i] | ((u32)c8[i] << 16);
            u32* dst = BU + kp * 64 + nc * 8;
            *(uint4*)(dst)     = *(uint4*)(o);
            *(uint4*)(dst + 4) = *(uint4*)(o + 4);
        }
    }
    __syncthreads();

    const int lane  = t & 31;
    const int wid   = t >> 5;
    const int waveM = wid & 1;          // 2 M-waves x 4 N-waves
    const int waveN = wid >> 1;
    const int half  = lane >> 4;
    const int l15   = lane & 15;
    const int mrow  = waveM * 32 + l15;       // A fragment 0 row; fragment 1 = +16
    const int ncol  = waveN * 16 + l15;

    union { u32 u[8]; bf16x16 v; } fa0, fa1, fb;
    union { float f[8]; f32x8 v; } acc0, acc1;
    #pragma unroll
    for (int i = 0; i < 8; i++) { acc0.f[i] = 0.f; acc1.f[i] = 0.f; }

    const u32* Arow0 = AU + mrow * K2p;
    const u32* Arow1 = Arow0 + 16 * K2p;
    const int  nkk   = K >> 5;
    for (int kk = 0; kk < nkk; kk++) {
        const int base = kk * 16;
        // A fragment 0 (rows mrow), A fragment 1 (rows mrow+16): adjacent-dword
        // pairs fuse to ds_load_2addr with contiguous destinations
        #pragma unroll
        for (int v = 0; v < 4; v++) fa0.u[v]     = Arow0[base + half * 4 + v];
        #pragma unroll
        for (int v = 0; v < 4; v++) fa0.u[4 + v] = Arow0[base + 8 + half * 4 + v];
        #pragma unroll
        for (int v = 0; v < 4; v++) fa1.u[v]     = Arow1[base + half * 4 + v];
        #pragma unroll
        for (int v = 0; v < 4; v++) fa1.u[4 + v] = Arow1[base + 8 + half * 4 + v];
        // single shared B fragment (stride-64 pairs fuse)
        const u32* brow = BU + (base + half * 4) * 64 + ncol;
        #pragma unroll
        for (int v = 0; v < 8; v++) fb.u[v] = brow[(v + (v >> 2) * 4) * 64];
#if defined(__HIP_DEVICE_COMPILE__)
        acc0.v = __builtin_amdgcn_wmma_f32_16x16x32_bf16(
            false, fa0.v, false, fb.v, (short)0, acc0.v, false, false);
        acc1.v = __builtin_amdgcn_wmma_f32_16x16x32_bf16(
            false, fa1.v, false, fb.v, (short)0, acc1.v, false, false);
#endif
    }

    // ---- epilogue: D layout = VGPR r -> M = r + 8*half, N = lane&15 ----
    const int mb0 = m0 + waveM * 32 + half * 8;
    const int ng  = n0 + waveN * 16 + l15;
    #pragma unroll
    for (int sub = 0; sub < 2; sub++) {
        const int    mbase = mb0 + sub * 16;
        const float* af    = sub ? acc1.f : acc0.f;
        #pragma unroll
        for (int r = 0; r < 8; r++) {
            int   m   = mbase + r;
            float val = af[r];
            if (EPI == EPI_RELU_F32) {
                val = fmaxf(val + bias[m], 0.f);
                ((float*)Y)[b * strideY + (long)m * ldy + ng] = val;
            } else if (EPI == EPI_RELU_BF16) {
                val = fmaxf(val + bias[m], 0.f);
                ((u16*)Y)[b * strideY + (long)m * ldy + ng] = f2bf(val);
            } else if (EPI == EPI_PLAIN_F32) {
                ((float*)Y)[b * strideY + (long)m * ldy + ng] = val;
            } else {  // EPI_GATE: G = sigmoid(conv(diff)); out1=pre0*G, out2=post0*(1-G)
                float g  = 1.f / (1.f + __expf(-(val + bias[m])));
                long idx = b * strideY + (long)m * ldy + ng;
                ((float*)Y)[idx] = pre0[idx] * g;
                Y2[idx]          = post0[idx] * (1.f - g);
            }
        }
    }
}

// ---------------------------------------------------------------------------
// Pooled queries: 10x10 uniform mean pool on 120x120, crop [1:-1] -> 8x8=64
// tokens of 12x12 means. Output [b][token][d] bf16 (A-operand layout).
// qkv rows 0..63 hold q.
// ---------------------------------------------------------------------------
__global__ void pool_q(const u16* __restrict__ qkv, u16* __restrict__ qtok)
{
    int d = threadIdx.x;           // 64
    int tk = blockIdx.x;           // 64 tokens
    int b = blockIdx.y;
    int ty = tk >> 3, tx = tk & 7;
    int y0 = (ty + 1) * 12, x0 = (tx + 1) * 12;
    const u16* q = qkv + ((long)b * 192 + d) * HW;
    float s = 0.f;
    for (int dy = 0; dy < 12; dy++) {
        const u16* row = q + (y0 + dy) * 120 + x0;
        #pragma unroll
        for (int dx = 0; dx < 12; dx++) s += bf2f(row[dx]);
    }
    qtok[((long)b * 64 + tk) * 64 + d] = f2bf(s * (1.f / 144.f));
}

// ---------------------------------------------------------------------------
// Per (b, token): softmax both logit rows over hw, then
// diff[b,i,n] = | v_post[i,n]*a_post[i,n] - v_pre[i,n]*a_pre[i,n] |  (bf16)
// ---------------------------------------------------------------------------
__global__ __launch_bounds__(256) void softmax_diff(
    const float* __restrict__ Lpre, const float* __restrict__ Lpost,
    const u16* __restrict__ qkv_pre, const u16* __restrict__ qkv_post,
    u16* __restrict__ diff)
{
    __shared__ float r1[256];
    __shared__ float r2[256];
    int i = blockIdx.x, b = blockIdx.y, t = threadIdx.x;
    const float* lp = Lpre  + ((long)b * 64 + i) * HW;
    const float* lq = Lpost + ((long)b * 64 + i) * HW;
    const u16*   vp = qkv_pre  + ((long)b * 192 + 128 + i) * HW;
    const u16*   vq = qkv_post + ((long)b * 192 + 128 + i) * HW;
    u16*         dd = diff + ((long)b * 64 + i) * HW;

    float mp = -3.4e38f, mq = -3.4e38f;
    for (int n = t; n < HW; n += 256) { mp = fmaxf(mp, lp[n]); mq = fmaxf(mq, lq[n]); }
    r1[t] = mp; r2[t] = mq; __syncthreads();
    for (int s = 128; s > 0; s >>= 1) {
        if (t < s) { r1[t] = fmaxf(r1[t], r1[t + s]); r2[t] = fmaxf(r2[t], r2[t + s]); }
        __syncthreads();
    }
    mp = r1[0]; mq = r2[0]; __syncthreads();

    float sp = 0.f, sq = 0.f;
    for (int n = t; n < HW; n += 256) { sp += __expf(lp[n] - mp); sq += __expf(lq[n] - mq); }
    r1[t] = sp; r2[t] = sq; __syncthreads();
    for (int s = 128; s > 0; s >>= 1) {
        if (t < s) { r1[t] += r1[t + s]; r2[t] += r2[t + s]; }
        __syncthreads();
    }
    float rp = 1.f / r1[0], rq = 1.f / r2[0];

    for (int n = t; n < HW; n += 256) {
        float ap = __expf(lp[n] - mp) * rp;
        float aq = __expf(lq[n] - mq) * rq;
        dd[n] = f2bf(fabsf(bf2f(vq[n]) * aq - bf2f(vp[n]) * ap));
    }
}

// ---------------------------------------------------------------------------
extern "C" void kernel_launch(void* const* d_in, const int* in_sizes, int n_in,
                              void* d_out, int out_size, void* d_ws, size_t ws_size,
                              hipStream_t stream)
{
    const float* pre   = (const float*)d_in[0];
    const float* post  = (const float*)d_in[1];
    const float* w_pp  = (const float*)d_in[2];
    const float* b_pp  = (const float*)d_in[3];
    const float* w_q   = (const float*)d_in[4];
    const float* b_q   = (const float*)d_in[5];
    const float* w_k   = (const float*)d_in[6];
    const float* b_k   = (const float*)d_in[7];
    const float* w_v   = (const float*)d_in[8];
    const float* b_v   = (const float*)d_in[9];
    const float* w_out = (const float*)d_in[10];
    const float* b_out = (const float*)d_in[11];
    const float* g_pp  = (const float*)d_in[12];
    const float* bt_pp = (const float*)d_in[13];
    const float* m_pp  = (const float*)d_in[14];
    const float* v_pp  = (const float*)d_in[15];
    const float* g_q   = (const float*)d_in[16];
    const float* bt_q  = (const float*)d_in[17];
    const float* m_q   = (const float*)d_in[18];
    const float* v_q   = (const float*)d_in[19];
    const float* g_k   = (const float*)d_in[20];
    const float* bt_k  = (const float*)d_in[21];
    const float* m_k   = (const float*)d_in[22];
    const float* v_k   = (const float*)d_in[23];
    const float* g_v   = (const float*)d_in[24];
    const float* bt_v  = (const float*)d_in[25];
    const float* m_v   = (const float*)d_in[26];
    const float* v_v   = (const float*)d_in[27];

    const long CHW = (long)CC * HW;        // 3,686,400 per batch
    char* w = (char*)d_ws;
    size_t off = 0;
    auto alloc = [&](size_t bytes) {
        void* p = w + off;
        off += (bytes + 255) & ~(size_t)255;
        return p;
    };
    u16*   preb    = (u16*)alloc((size_t)NB * CHW * 2);        // later reused as logits_pre (f32, half size)
    u16*   postb   = (u16*)alloc((size_t)NB * CHW * 2);        // later reused as logits_post
    float* pre0    = (float*)alloc((size_t)NB * CHW * 4);
    float* post0   = (float*)alloc((size_t)NB * CHW * 4);
    u16*   qkvpre  = (u16*)alloc((size_t)NB * 192 * HW * 2);
    u16*   qkvpost = (u16*)alloc((size_t)NB * 192 * HW * 2);
    u16*   diff    = (u16*)alloc((size_t)NB * 64 * HW * 2);
    u16*   qtokpre = (u16*)alloc((size_t)NB * 64 * 64 * 2);
    u16*   qtokpost= (u16*)alloc((size_t)NB * 64 * 64 * 2);
    u16*   wpp     = (u16*)alloc(65536 * 2);
    u16*   wqkv    = (u16*)alloc(49152 * 2);
    u16*   wout    = (u16*)alloc(16384 * 2);
    float* bpp     = (float*)alloc(256 * 4);
    float* bqkv    = (float*)alloc(192 * 4);
    float* bout    = (float*)alloc(256 * 4);
    float* logits_pre  = (float*)preb;     // alias: bf16 inputs dead by then
    float* logits_post = (float*)postb;

    // 1) fold BN into weights, convert weights to bf16
    prep_kernel<<<512, 256, 0, stream>>>(
        w_pp, b_pp, g_pp, bt_pp, m_pp, v_pp,
        w_q, b_q, g_q, bt_q, m_q, v_q,
        w_k, b_k, g_k, bt_k, m_k, v_k,
        w_v, b_v, g_v, bt_v, m_v, v_v,
        w_out, b_out, wpp, bpp, wqkv, bqkv, wout, bout);

    // 2) activations -> bf16
    const int n4 = (int)(NB * CHW / 4);    // 7,372,800
    cvt_bf16<<<n4 / 256, 256, 0, stream>>>(pre,  preb,  n4);
    cvt_bf16<<<n4 / 256, 256, 0, stream>>>(post, postb, n4);

    // 3) pre_0 / post_0 = relu(BN-folded conv_pp)   [M=256,K=256,N=HW per b]
    gemm_wmma<EPI_RELU_F32><<<dim3(HW / 64, 4, NB), 256, 0, stream>>>(
        wpp, 0, preb,  CHW, HW, bpp, pre0,  CHW, HW, 256, nullptr, nullptr, nullptr);
    gemm_wmma<EPI_RELU_F32><<<dim3(HW / 64, 4, NB), 256, 0, stream>>>(
        wpp, 0, postb, CHW, HW, bpp, post0, CHW, HW, 256, nullptr, nullptr, nullptr);

    // 4) fused q/k/v maps (bf16)   [M=192,K=256,N=HW per b]
    gemm_wmma<EPI_RELU_BF16><<<dim3(HW / 64, 3, NB), 256, 0, stream>>>(
        wqkv, 0, preb,  CHW, HW, bqkv, qkvpre,  (long)192 * HW, HW, 256, nullptr, nullptr, nullptr);
    gemm_wmma<EPI_RELU_BF16><<<dim3(HW / 64, 3, NB), 256, 0, stream>>>(
        wqkv, 0, postb, CHW, HW, bqkv, qkvpost, (long)192 * HW, HW, 256, nullptr, nullptr, nullptr);

    // 5) pooled query tokens
    pool_q<<<dim3(64, NB), 64, 0, stream>>>(qkvpre,  qtokpre);
    pool_q<<<dim3(64, NB), 64, 0, stream>>>(qkvpost, qtokpost);

    // 6) attention logits via WMMA: Q[64,64] x K[64,HW]   (cross: pre_q . post_k)
    gemm_wmma<EPI_PLAIN_F32><<<dim3(HW / 64, 1, NB), 256, 0, stream>>>(
        qtokpre,  4096, qkvpost + (long)64 * HW, (long)192 * HW, HW, nullptr,
        logits_pre,  (long)64 * HW, HW, 64, nullptr, nullptr, nullptr);
    gemm_wmma<EPI_PLAIN_F32><<<dim3(HW / 64, 1, NB), 256, 0, stream>>>(
        qtokpost, 4096, qkvpre  + (long)64 * HW, (long)192 * HW, HW, nullptr,
        logits_post, (long)64 * HW, HW, 64, nullptr, nullptr, nullptr);

    // 7) softmax + |post_x - pre_x|  -> diff (bf16)
    softmax_diff<<<dim3(64, NB), 256, 0, stream>>>(
        logits_pre, logits_post, qkvpre, qkvpost, diff);

    // 8) G = sigmoid(conv_out(diff)); out1 = pre_0*G, out2 = post_0*(1-G)
    float* out1 = (float*)d_out;
    float* out2 = out1 + (size_t)NB * CHW;
    gemm_wmma<EPI_GATE><<<dim3(HW / 64, 4, NB), 256, 0, stream>>>(
        wout, 0, diff, (long)64 * HW, HW, bout, out1, CHW, HW, 64, pre0, post0, out2);
}
